// Net_1417339207784
// MI455X (gfx1250) — matrix operand order repaired and checked by reading, not compiled, not run on hardware
//
#include <hip/hip_runtime.h>
#include <hip/hip_bf16.h>

typedef __attribute__((ext_vector_type(16))) _Float16 v16h;
typedef __attribute__((ext_vector_type(8)))  _Float16 v8h;
typedef __attribute__((ext_vector_type(8)))  float    v8f;

#define Hd 128
#define Wd 128
#define HWp 16384            // H*W
#define Bn 2
#define DM_ 64
#define DI_ 128
#define DS_ 16
#define L_TOK 32768          // B*H*W
#define TCHUNK 256
#define NCHUNK 128           // L_TOK / TCHUNK

// ---------------- helpers ----------------
__device__ __forceinline__ float sigm_(float x) { return 1.f / (1.f + __expf(-x)); }

// act codes: 0 none, 1 lrelu(0.1), 2 lrelu(0.2), 3 sigmoid, 6 softplus
__device__ __forceinline__ float apply_act(float v, int act) {
  switch (act) {
    case 1: return v >= 0.f ? v : 0.1f * v;
    case 2: return v >= 0.f ? v : 0.2f * v;
    case 3: return sigm_(v);
    case 6: return v > 20.f ? v : log1pf(__expf(v));
    default: return v;
  }
}

// A-fragment (16x32 f16) K index for element e, lane-half hi  (ISA 7.12.2)
__device__ __forceinline__ int a_kidx(int e, int hi) {
  int v = e >> 1, p = e & 1;
  int k = (v < 4) ? (2 * v) : (2 * v + 8);
  return k + (hi ? 8 : 0) + p;
}

__device__ __forceinline__ int iclamp(int v, int lo, int hi) {
  return v < lo ? lo : (v > hi ? hi : v);
}

// ---------------- channel-slice copy (builds concat buffers) ----------------
__global__ void copy_chan(const float* __restrict__ src, long srcBStride, int srcC0,
                          float* __restrict__ dst, int dstC, int dstC0, int nC) {
  long i = (long)blockIdx.x * blockDim.x + threadIdx.x;
  long tot = (long)Bn * nC * HWp;
  if (i >= tot) return;
  int hw = (int)(i & (HWp - 1));
  int c  = (int)((i >> 14) % nC);
  int b  = (int)(i / ((long)nC * HWp));
  dst[((long)(b * dstC + dstC0 + c) << 14) + hw] =
      src[(long)b * srcBStride + ((long)(srcC0 + c) << 14) + hw];
}

// ---------------- 3x3 conv (pad=1) as implicit GEMM via WMMA f16->f32 ----------------
// Block = 4 waves (128 threads), one full 128-pixel output row, one 16-channel co-tile.
//  * Weights staged once in LDS in WMMA B-fragment order -> per-lane v16h (2x ds_load_b128).
//  * Per 32-channel K-chunk, a 3-row x 130-pixel input slab is staged in LDS as f16 with
//    halo zeros; the per-lane A K-pattern is two contiguous 8-half runs -> 2x ds_load_b128.
//  * Each wave computes two 16-pixel M-tiles sharing each B fragment (2 WMMAs/frag).
// grid: (H, Cout/16, B), block: 128. Cin must be a multiple of 32 (max 128).
__global__ void conv3x3_wmma(const float* __restrict__ in, long inBStride,
                             const float* __restrict__ w,     // (Cout,Cin,3,3)
                             const float* __restrict__ bias,  // (Cout) or null
                             const float* __restrict__ resid, // NCHW or null (added after act)
                             float* __restrict__ out,         // (B,Cout,H,W)
                             int Cin, int Cout, int act) {
  __shared__ __attribute__((aligned(32))) _Float16 wsh[128 * 9 * 16]; // frag-ordered weights
  __shared__ __attribute__((aligned(32))) _Float16 ish[3 * 132 * 32]; // input slab (1 K-chunk)

  const int tid  = threadIdx.x;
  const int lane = tid & 31;
  const int wid  = tid >> 5;
  const int hi = lane >> 4;
  const int ln = lane & 15;
  const int h   = blockIdx.x;
  const int w0  = wid << 5;                 // 32-pixel strip per wave
  const int cob = blockIdx.y << 4;
  const int b   = blockIdx.z;
  const int CC  = Cin >> 5;                 // # of 32-wide K chunks

  // stage weights in fragment order:
  // wsh[((khkw*CC+cc)<<9) + l*16 + e] = w[(cob + (l&15))*Cin*9 + (cc*32 + e + 16*(l>>4))*9 + khkw]
  const int wtot = CC * 9 * 512;            // == Cin*9*16
  for (int j = tid; j < wtot; j += 128) {
    int e = j & 15;
    int l = (j >> 4) & 31;
    int rest = j >> 9;                      // khkw*CC + cc
    int cc = rest % CC;
    int khkw = rest / CC;
    int k = cc * 32 + e + ((l >> 4) << 4);
    wsh[j] = (_Float16)w[(long)(cob + (l & 15)) * Cin * 9 + k * 9 + khkw];
  }

  const float* inb = in + (long)b * inBStride;
  v8f acc0 = {}, acc1 = {};
  for (int cib = 0; cib < Cin; cib += 32) {
    const int cc = cib >> 5;
    __syncthreads();                        // prior reads done (also covers weight staging)
    // stage input slab: ish[((row*132 + lpix)<<5) + ci] = in[cib+ci][h+row-1][lpix-1] (0 on halo)
    for (int j = tid; j < 3 * 32 * 132; j += 128) {
      int lpix = j % 132;
      int t = j / 132;
      int ci = t & 31;
      int row = t >> 5;
      int ih = h + row - 1;
      int iw = lpix - 1;
      float okf = (ih >= 0 && ih < Hd && iw >= 0 && iw < Wd) ? 1.f : 0.f;
      long gidx = ((long)(cib + ci) << 14) + (long)iclamp(ih, 0, Hd - 1) * Wd + iclamp(iw, 0, Wd - 1);
      ish[((row * 132 + lpix) << 5) + ci] = (_Float16)(inb[gidx] * okf);
    }
    __syncthreads();

    for (int kh = 0; kh < 3; ++kh) {
      for (int kw = 0; kw < 3; ++kw) {
        int khkw = kh * 3 + kw;
        // B fragment: one contiguous 32B read per lane
        v16h bf = *(const v16h*)&wsh[(((khkw)*CC + cc) << 9) + (lane << 4)];
        // A fragments: one contiguous 16-half run per lane (K = 16*hi + 0..15)
        int lpix0 = w0 + ln + kw;           // (iw+1) for tile0
        const _Float16* ap0 = &ish[((kh * 132 + lpix0) << 5) + (hi << 4)];
        const _Float16* ap1 = ap0 + (16 << 5);   // tile1: +16 pixels
        v8h a0lo = *(const v8h*)ap0;
        v8h a0hi = *(const v8h*)(ap0 + 8);
        v8h a1lo = *(const v8h*)ap1;
        v8h a1hi = *(const v8h*)(ap1 + 8);
        v16h a0 = __builtin_shufflevector(a0lo, a0hi, 0,1,2,3,4,5,6,7,8,9,10,11,12,13,14,15);
        v16h a1 = __builtin_shufflevector(a1lo, a1hi, 0,1,2,3,4,5,6,7,8,9,10,11,12,13,14,15);
        acc0 = __builtin_amdgcn_wmma_f32_16x16x32_f16(false, a0, false, bf,
                                                      (short)0, acc0, false, false);
        acc1 = __builtin_amdgcn_wmma_f32_16x16x32_f16(false, a1, false, bf,
                                                      (short)0, acc1, false, false);
      }
    }
  }
  float bv = bias ? bias[cob + ln] : 0.f;
  long obase = ((long)(b * Cout + cob + ln) << 14) + (long)h * Wd + w0;
#pragma unroll
  for (int v = 0; v < 8; ++v) {
    int m = v + (hi ? 8 : 0);                 // C/D: M = v + 8*laneHi
    long o0 = obase + m;
    long o1 = obase + 16 + m;
    float v0 = apply_act(acc0[v] + bv, act);
    float v1 = apply_act(acc1[v] + bv, act);
    if (resid) { v0 += resid[o0]; v1 += resid[o1]; }
    out[o0] = v0;
    out[o1] = v1;
  }
}

// ---------------- generic strided GEMM: C[m,n] = sum_k A[m,k]*B[k,n] ----------------
// Two 16-row M-tiles per wave share one B fragment. M must be a multiple of 32.
// grid: (M/32, ceil(N/16)), block 32.
__global__ void gemm_wmma(const float* __restrict__ A, long sAm, long sAk,
                          const float* __restrict__ Bw, long sBk, long sBn,
                          float* __restrict__ Co, long sCm, long sCn,
                          int M, int N, int K,
                          const float* __restrict__ bias, int act) {
  const int lane = threadIdx.x;
  const int hi = lane >> 4;
  const int ln = lane & 15;
  const int mb = blockIdx.x << 5;
  const int nb = blockIdx.y << 4;
  const int m0 = mb + ln;
  const int m1 = mb + 16 + ln;
  const int n  = nb + ln;
  const int nC = iclamp(n, 0, N - 1);
  const float nok = (n < N) ? 1.f : 0.f;
  v8f acc0 = {}, acc1 = {};
  for (int kb = 0; kb < K; kb += 32) {
    v16h a0 = {}, a1 = {}, bf = {};
#pragma unroll
    for (int e = 0; e < 16; ++e) {
      int ka = kb + a_kidx(e, hi);
      int kaC = iclamp(ka, 0, K - 1);
      float kok = (ka < K) ? 1.f : 0.f;
      a0[e] = (_Float16)(A[(long)m0 * sAm + (long)kaC * sAk] * kok);
      a1[e] = (_Float16)(A[(long)m1 * sAm + (long)kaC * sAk] * kok);
      int kk = kb + a_kidx(e, hi);
      int kkC = iclamp(kk, 0, K - 1);
      float bok = (kk < K) ? nok : 0.f;
      bf[e] = (_Float16)(Bw[(long)kkC * sBk + (long)nC * sBn] * bok);
    }
    acc0 = __builtin_amdgcn_wmma_f32_16x16x32_f16(false, a0, false, bf,
                                                  (short)0, acc0, false, false);
    acc1 = __builtin_amdgcn_wmma_f32_16x16x32_f16(false, a1, false, bf,
                                                  (short)0, acc1, false, false);
  }
  float bv = (bias && n < N) ? bias[n] : 0.f;
#pragma unroll
  for (int v = 0; v < 8; ++v) {
    int mr0 = mb + v + (hi ? 8 : 0);
    int mr1 = mr0 + 16;
    if (n < N) {
      Co[(long)mr0 * sCm + (long)n * sCn] = apply_act(acc0[v] + bv, act);
      Co[(long)mr1 * sCm + (long)n * sCn] = apply_act(acc1[v] + bv, act);
    }
  }
}

// ---------------- causal depthwise conv1d (DCONV=4) + bias + SiLU ----------------
// xz: (L,256), first 128 cols = xm; out: (L,128)
__global__ void dwconv_silu(const float* __restrict__ xz, const float* __restrict__ w,
                            const float* __restrict__ bias, float* __restrict__ out) {
  long i = (long)blockIdx.x * blockDim.x + threadIdx.x;
  if (i >= (long)L_TOK * DI_) return;
  int d = (int)(i & (DI_ - 1));
  long l = i >> 7;
  float s = bias[d];
#pragma unroll
  for (int j = 0; j < 4; ++j) {
    long l2 = l - 3 + j;
    if (l2 >= 0) s += xz[l2 * 256 + d] * w[d * 4 + j];
  }
  out[i] = s * sigm_(s);
}

// ---------------- selective scan: 3-pass chunked associative scan ----------------
// lane mapping: s = lane&15, d = blockIdx.y*2 + (lane>>4)
__global__ void scan_passA(const float* __restrict__ delta, const float* __restrict__ dbc,
                           const float* __restrict__ u, const float* __restrict__ A_log,
                           float* __restrict__ cA, float* __restrict__ cB) {
  const int lane = threadIdx.x;
  const int s = lane & 15;
  const int d = (blockIdx.y << 1) + (lane >> 4);
  const float Av = -__expf(A_log[d * DS_ + s]);
  long l0 = (long)blockIdx.x * TCHUNK;
  float accA = 1.f, accB = 0.f;
  for (int i = 0; i < TCHUNK; ++i) {
    long l = l0 + i;
    float dl = delta[l * DI_ + d];
    float bm = dbc[l * 36 + 4 + s];
    float uu = u[l * DI_ + d];
    float a = __expf(dl * Av);
    accB = a * accB + dl * bm * uu;
    accA *= a;
  }
  long idx = ((long)blockIdx.x * DI_ + d) * DS_ + s;
  cA[idx] = accA; cB[idx] = accB;
}

__global__ void scan_passB(const float* __restrict__ cA, const float* __restrict__ cB,
                           float* __restrict__ hinit) {
  const int lane = threadIdx.x;
  const int s = lane & 15;
  const int d = (blockIdx.x << 1) + (lane >> 4);
  float h = 0.f;
  for (int c = 0; c < NCHUNK; ++c) {
    long idx = ((long)c * DI_ + d) * DS_ + s;
    hinit[idx] = h;
    h = cA[idx] * h + cB[idx];
  }
}

__global__ void scan_passC(const float* __restrict__ delta, const float* __restrict__ dbc,
                           const float* __restrict__ u, const float* __restrict__ A_log,
                           const float* __restrict__ Dp, const float* __restrict__ hinit,
                           float* __restrict__ y) {
  const int lane = threadIdx.x;
  const int s = lane & 15;
  const int d = (blockIdx.y << 1) + (lane >> 4);
  const float Av = -__expf(A_log[d * DS_ + s]);
  const float Dv = Dp[d];
  long l0 = (long)blockIdx.x * TCHUNK;
  float h = hinit[((long)blockIdx.x * DI_ + d) * DS_ + s];
  for (int i = 0; i < TCHUNK; ++i) {
    long l = l0 + i;
    float dl = delta[l * DI_ + d];
    float bm = dbc[l * 36 + 4 + s];
    float cm = dbc[l * 36 + 20 + s];
    float uu = u[l * DI_ + d];
    float a = __expf(dl * Av);
    h = a * h + dl * bm * uu;
    float contrib = h * cm;
#pragma unroll
    for (int o = 8; o > 0; o >>= 1) contrib += __shfl_xor(contrib, o, 16);
    if (s == 0) y[l * DI_ + d] = contrib + uu * Dv;
  }
}

// y *= silu(z)   (z = xz cols [128,256))
__global__ void gate_mul(float* __restrict__ y, const float* __restrict__ xz) {
  long i = (long)blockIdx.x * blockDim.x + threadIdx.x;
  if (i >= (long)L_TOK * DI_) return;
  int d = (int)(i & (DI_ - 1));
  long l = i >> 7;
  float z = xz[l * 256 + 128 + d];
  y[i] *= z * sigm_(z);
}

// BN + lrelu(0.2) + residual, token-major (L,64) -> NCHW
__global__ void bn_lrelu_res(const float* __restrict__ ym, const float* __restrict__ gamma,
                             const float* __restrict__ beta, const float* __restrict__ mean,
                             const float* __restrict__ var, const float* __restrict__ res,
                             float* __restrict__ out) {
  long i = (long)blockIdx.x * blockDim.x + threadIdx.x;
  if (i >= (long)Bn * DM_ * HWp) return;
  int bc = (int)(i >> 14);
  int c = bc & 63, b = bc >> 6;
  int hw = (int)(i & (HWp - 1));
  float v = ym[((long)b * HWp + hw) * DM_ + c];
  v = (v - mean[c]) * (gamma[c] * rsqrtf(var[c] + 1e-5f)) + beta[c];
  out[i] = (v >= 0.f ? v : 0.2f * v) + res[i];
}

// per-(b,c) mean & max over HxW
__global__ void reduce_meanmax(const float* __restrict__ in, float* __restrict__ avg,
                               float* __restrict__ mx) {
  __shared__ float ssum[256], smax[256];
  int t = threadIdx.x, blk = blockIdx.x;
  const float* p = in + ((long)blk << 14);
  float s = 0.f, m = -3.4e38f;
  for (int i = t; i < HWp; i += 256) { float v = p[i]; s += v; m = fmaxf(m, v); }
  ssum[t] = s; smax[t] = m; __syncthreads();
  for (int o = 128; o > 0; o >>= 1) {
    if (t < o) { ssum[t] += ssum[t + o]; smax[t] = fmaxf(smax[t], smax[t + o]); }
    __syncthreads();
  }
  if (t == 0) { avg[blk] = ssum[0] * (1.f / HWp); mx[blk] = smax[0]; }
}

// channel-attention MLP: sigmoid(relu(avg@W1^T)@W2^T + relu(mx@W1^T)@W2^T), 1 block of 128
__global__ void channel_fc(const float* __restrict__ avg, const float* __restrict__ mx,
                           const float* __restrict__ fc1, const float* __restrict__ fc2,
                           float* __restrict__ ca) {
  int t = threadIdx.x;              // 0..127  -> (b, cout)
  int b = t >> 6, co = t & 63;
  float acc = 0.f;
  for (int h = 0; h < 16; ++h) {
    float ha = 0.f, hm = 0.f;
    for (int c = 0; c < 64; ++c) {
      float w1 = fc1[h * 64 + c];
      ha += w1 * avg[b * 64 + c];
      hm += w1 * mx[b * 64 + c];
    }
    acc += fc2[co * 16 + h] * (fmaxf(ha, 0.f) + fmaxf(hm, 0.f));
  }
  ca[t] = sigm_(acc);
}

__global__ void scale_ch(const float* __restrict__ in, const float* __restrict__ ca,
                         float* __restrict__ out) {
  long i = (long)blockIdx.x * blockDim.x + threadIdx.x;
  if (i >= (long)Bn * DM_ * HWp) return;
  out[i] = ca[i >> 14] * in[i];
}

// out = saA*tB + ca*sa*f0
__global__ void final_combine(const float* __restrict__ saA, const float* __restrict__ tB,
                              const float* __restrict__ ca, const float* __restrict__ sa,
                              const float* __restrict__ f0, float* __restrict__ out) {
  long i = (long)blockIdx.x * blockDim.x + threadIdx.x;
  if (i >= (long)Bn * DM_ * HWp) return;
  out[i] = saA[i] * tB[i] + ca[i >> 14] * sa[i] * f0[i];
}

// ---------------- host launcher ----------------
extern "C" void kernel_launch(void* const* d_in, const int* in_sizes, int n_in,
                              void* d_out, int out_size, void* d_ws, size_t ws_size,
                              hipStream_t stream) {
  (void)in_sizes; (void)n_in; (void)out_size; (void)ws_size;
  auto P = [&](int i) { return (const float*)d_in[i]; };
  const float* x         = P(0);
  const float* conv1_w   = P(1);  const float* conv1_b  = P(2);
  const float* conv2_w   = P(3);  const float* conv2_b  = P(4);
  const float* conv_w    = P(5);  const float* conv_b   = P(6);
  const float* sa1_w     = P(7);  const float* sa1_b    = P(8);
  const float* sa2_w     = P(9);  const float* sa2_b    = P(10);
  const float* ca_fc1    = P(11); const float* ca_fc2   = P(12);
  const float* rc1_w     = P(13); const float* rc1_b    = P(14);
  const float* rc2_w     = P(15); const float* rc2_b    = P(16);
  const float* bn_gamma  = P(17); const float* bn_beta  = P(18);
  const float* bn_mean   = P(19); const float* bn_var   = P(20);
  const float* in_proj_w = P(21);
  const float* conv1d_w  = P(22); const float* conv1d_b = P(23);
  const float* x_proj_w  = P(24);
  const float* dt_proj_w = P(25); const float* dt_proj_b= P(26);
  const float* A_log     = P(27); const float* Dp       = P(28);
  const float* out_proj_w= P(29);
  const float* agx_w     = P(30); const float* agx_b    = P(31);
  const float* ag_fc1    = P(32); const float* ag_fc2   = P(33);
  const float* ag_sa1_w  = P(34); const float* ag_sa1_b = P(35);
  const float* ag_sa2_w  = P(36); const float* ag_sa2_b = P(37);
  float* out = (float*)d_out;

  size_t off = 0;
  auto alloc = [&](size_t nflt) -> float* {
    float* p = (float*)((char*)d_ws + off);
    off += ((nflt * sizeof(float) + 255) & ~(size_t)255);
    return p;
  };
  const long nBC = (long)Bn * DM_ * HWp;          // 2^21
  float* spa_in = alloc((long)Bn * 96 * HWp);
  float* spe_in = alloc((long)Bn * 96 * HWp);
  float* spa0   = alloc(nBC);
  float* spe0   = alloc(nBC);
  float* spa    = alloc(nBC);
  float* spe    = alloc(nBC);
  float* f0     = alloc(nBC);
  float* sa_buf = alloc(nBC);
  float* saA    = alloc(nBC);
  float* t32    = alloc((long)Bn * 32 * HWp);
  float* t128   = alloc((long)Bn * 128 * HWp);
  float* xcat   = alloc((long)Bn * 128 * HWp);
  float* tA     = alloc(nBC);
  float* tB     = alloc(nBC);
  float* xz     = alloc((long)L_TOK * 256);
  float* xm_act = alloc((long)L_TOK * DI_);
  float* dbc    = alloc((long)L_TOK * 36);
  float* delta  = alloc((long)L_TOK * DI_);
  float* ybuf   = alloc((long)L_TOK * DI_);
  float* ym     = alloc((long)L_TOK * DM_);
  float* cA     = alloc((long)NCHUNK * DI_ * DS_);
  float* cB     = alloc((long)NCHUNK * DI_ * DS_);
  float* hinit  = alloc((long)NCHUNK * DI_ * DS_);
  float* avg1   = alloc(128); float* mx1 = alloc(128); float* ca_buf = alloc(128);
  float* avg2   = alloc(128); float* mx2 = alloc(128); float* caA    = alloc(128);

  const long xBS = 128L * HWp;                  // batch stride of x (elements)
  auto cgrid = [](long n) { return dim3((unsigned)((n + 255) / 256)); };

  // concat inputs: spa_in=[msi,xm], spe_in=[hsi,xm]
  copy_chan<<<cgrid((long)Bn*32*HWp),256,0,stream>>>(x, xBS, 64, spa_in, 96, 0, 32);
  copy_chan<<<cgrid((long)Bn*64*HWp),256,0,stream>>>(x, xBS, 0,  spa_in, 96, 32, 64);
  copy_chan<<<cgrid((long)Bn*32*HWp),256,0,stream>>>(x, xBS, 96, spe_in, 96, 0, 32);
  copy_chan<<<cgrid((long)Bn*64*HWp),256,0,stream>>>(x, xBS, 0,  spe_in, 96, 32, 64);

  dim3 cg64(Hd, 4, Bn), cg128(Hd, 8, Bn), cg32(Hd, 2, Bn);   // one row per block
  // conv1 / conv2 (96 -> 64)
  conv3x3_wmma<<<cg64,128,0,stream>>>(spa_in, 96L*HWp, conv1_w, conv1_b, nullptr, spa0, 96, 64, 0);
  conv3x3_wmma<<<cg64,128,0,stream>>>(spe_in, 96L*HWp, conv2_w, conv2_b, nullptr, spe0, 96, 64, 0);

  // ---- mamba on spa0 ----
  for (int b = 0; b < Bn; ++b)   // in_proj: xf(L,64) @ W^T(64,256); A strides do the NCHW transpose
    gemm_wmma<<<dim3(512,16),32,0,stream>>>(spa0 + (long)b*DM_*HWp, 1, HWp,
        in_proj_w, 1, 64, xz + (long)b*HWp*256, 256, 1, HWp, 256, 64, nullptr, 0);
  dwconv_silu<<<cgrid((long)L_TOK*DI_),256,0,stream>>>(xz, conv1d_w, conv1d_b, xm_act);
  // x_proj: (L,128)@(128,36)
  gemm_wmma<<<dim3(1024,3),32,0,stream>>>(xm_act, 128, 1, x_proj_w, 1, 128,
                                          dbc, 36, 1, L_TOK, 36, 128, nullptr, 0);
  // dt_proj + softplus: (L,4)@(4,128)
  gemm_wmma<<<dim3(1024,8),32,0,stream>>>(dbc, 36, 1, dt_proj_w, 1, 4,
                                          delta, 128, 1, L_TOK, 128, 4, dt_proj_b, 6);
  scan_passA<<<dim3(NCHUNK, DI_/2),32,0,stream>>>(delta, dbc, xm_act, A_log, cA, cB);
  scan_passB<<<dim3(DI_/2),32,0,stream>>>(cA, cB, hinit);
  scan_passC<<<dim3(NCHUNK, DI_/2),32,0,stream>>>(delta, dbc, xm_act, A_log, Dp, hinit, ybuf);
  gate_mul<<<cgrid((long)L_TOK*DI_),256,0,stream>>>(ybuf, xz);
  // out_proj: (L,128)@(128,64)
  gemm_wmma<<<dim3(1024,4),32,0,stream>>>(ybuf, 128, 1, out_proj_w, 1, 128,
                                          ym, 64, 1, L_TOK, 64, 128, nullptr, 0);
  bn_lrelu_res<<<cgrid(nBC),256,0,stream>>>(ym, bn_gamma, bn_beta, bn_mean, bn_var, spa0, spa);

  // ---- spe branch: residual conv stack ----
  conv3x3_wmma<<<cg128,128,0,stream>>>(spe0, 64L*HWp, rc1_w, rc1_b, nullptr, t128, 64, 128, 2);
  conv3x3_wmma<<<cg64 ,128,0,stream>>>(t128, 128L*HWp, rc2_w, rc2_b, spe0,    spe,  128, 64, 2);

  // f0 = conv(xm)  (xm = x channels [0,64), batch stride 128*HW)
  conv3x3_wmma<<<cg64,128,0,stream>>>(x, xBS, conv_w, conv_b, nullptr, f0, 64, 64, 0);

  // sa = spatial_att(spa)
  conv3x3_wmma<<<cg32,128,0,stream>>>(spa, 64L*HWp, sa1_w, sa1_b, nullptr, t32, 64, 32, 1);
  conv3x3_wmma<<<cg64,128,0,stream>>>(t32, 32L*HWp, sa2_w, sa2_b, nullptr, sa_buf, 32, 64, 3);
  // ca = channel_att(spe)
  reduce_meanmax<<<128,256,0,stream>>>(spe, avg1, mx1);
  channel_fc<<<1,128,0,stream>>>(avg1, mx1, ca_fc1, ca_fc2, ca_buf);

  // ---- aglf(spe, spa) ----
  copy_chan<<<cgrid(nBC),256,0,stream>>>(spe, 64L*HWp, 0, xcat, 128, 0,  64);
  copy_chan<<<cgrid(nBC),256,0,stream>>>(spa, 64L*HWp, 0, xcat, 128, 64, 64);
  conv3x3_wmma<<<cg64,128,0,stream>>>(xcat, 128L*HWp, agx_w, agx_b, nullptr, tA, 128, 64, 0);
  reduce_meanmax<<<128,256,0,stream>>>(tA, avg2, mx2);
  channel_fc<<<1,128,0,stream>>>(avg2, mx2, ag_fc1, ag_fc2, caA);
  scale_ch<<<cgrid(nBC),256,0,stream>>>(tA, caA, tB);
  conv3x3_wmma<<<cg32,128,0,stream>>>(tB,  64L*HWp, ag_sa1_w, ag_sa1_b, nullptr, t32, 64, 32, 1);
  conv3x3_wmma<<<cg64,128,0,stream>>>(t32, 32L*HWp, ag_sa2_w, ag_sa2_b, nullptr, saA, 32, 64, 3);

  // out = saA*tB + ca*sa*f0
  final_combine<<<cgrid(nBC),256,0,stream>>>(saA, tB, ca_buf, sa_buf, f0, out);
}
